// EqvarLayer_42039139893968
// MI455X (gfx1250) — compile-verified, matrix operand order checked by reading, not run on hardware
//
#include <hip/hip_runtime.h>
#include <hip/hip_bf16.h>

typedef __attribute__((ext_vector_type(16))) _Float16 v16h;
typedef __attribute__((ext_vector_type(8)))  float    v8f;
typedef __attribute__((ext_vector_type(4)))  float    v4f;   // native vector (NT-store OK)

#define FD 64

// ---------------------------------------------------------------------------
// Kernel 1: fold the three 64x64 weight matrices into P_i = W_ii@W_i@W_pp and
// P_j = W_ii@W_j@W_pp (full fp32), then store f16 hi/lo splits for the WMMA
// node GEMM. Tiny (~2 MFLOP) -> one 256-thread block.
// ---------------------------------------------------------------------------
__global__ __launch_bounds__(256) void fold_weights_kernel(
    const float* __restrict__ W_pp, const float* __restrict__ W_i,
    const float* __restrict__ W_j,  const float* __restrict__ W_ii,
    _Float16* __restrict__ Phi_i, _Float16* __restrict__ Plo_i,
    _Float16* __restrict__ Phi_j, _Float16* __restrict__ Plo_j)
{
  __shared__ float T[FD * FD];
  const int t = threadIdx.x;
  for (int m = 0; m < 2; ++m) {
    const float* Wm = (m == 0) ? W_i : W_j;
    _Float16* Ph = (m == 0) ? Phi_i : Phi_j;
    _Float16* Pl = (m == 0) ? Plo_i : Plo_j;
    __syncthreads();
    // T = Wm @ W_pp : T[r][f] = sum_s Wm[r][s] * W_pp[s][f]
    for (int idx = t; idx < FD * FD; idx += 256) {
      const int r = idx >> 6, f = idx & 63;
      float acc = 0.f;
      for (int s = 0; s < FD; ++s) acc = fmaf(Wm[r * FD + s], W_pp[s * FD + f], acc);
      T[idx] = acc;
    }
    __syncthreads();
    // P = W_ii @ T, split into f16 hi + f16 lo (hi+lo ~ fp32 accurate)
    for (int idx = t; idx < FD * FD; idx += 256) {
      const int g = idx >> 6, f = idx & 63;
      float acc = 0.f;
      for (int r = 0; r < FD; ++r) acc = fmaf(W_ii[g * FD + r], T[r * FD + f], acc);
      const _Float16 h = (_Float16)acc;
      Ph[idx] = h;
      Pl[idx] = (_Float16)(acc - (float)h);
    }
  }
}

// ---------------------------------------------------------------------------
// WMMA operand helpers. ISA 7.12.2 layout for 16-bit 16x32 A (wave32):
// lane = r + 16*hb holds row M=r; element e<8 -> K = 8*hb+e, e>=8 -> K = 16+8*hb+(e-8).
// B (32x16, K-major transposed weights) mirrors A with N in place of M.
// ---------------------------------------------------------------------------
__device__ __forceinline__ void make_frag_split_f32(const float* __restrict__ p,
                                                    int k0, int hb, v16h& fh, v16h& fl) {
#pragma unroll
  for (int e = 0; e < 16; ++e) {
    const int kl = (e < 8) ? (8 * hb + e) : (8 * hb + e + 8);
    const float x = p[k0 + kl];
    const _Float16 xh = (_Float16)x;
    fh[e] = xh;
    fl[e] = (_Float16)(x - (float)xh);
  }
}

__device__ __forceinline__ void load_frag_f16(const _Float16* __restrict__ p,
                                              int k0, int hb, v16h& f) {
#pragma unroll
  for (int e = 0; e < 16; ++e) {
    const int kl = (e < 8) ? (8 * hb + e) : (8 * hb + e + 8);
    f[e] = p[k0 + kl];
  }
}

#define WMMA_F16(A, B, C) \
  __builtin_amdgcn_wmma_f32_16x16x32_f16(false, (A), false, (B), (short)0, (C), false, false)

// ---------------------------------------------------------------------------
// Kernel 2: node GEMM, FULL 16-row tiles only (no guards -> straight-line
// stores, EXEC all-ones throughout). y_i = px @ P_i^T, y_j = px @ P_j^T.
// 8 waves/block, one tile per wave, K=64 via two k-steps, hi/lo split with
// two independent accumulator chains to break WMMA->WMMA RAW latency.
// ---------------------------------------------------------------------------
__global__ __launch_bounds__(256) void node_gemm_kernel(
    const float* __restrict__ px,
    const _Float16* __restrict__ Phi_i, const _Float16* __restrict__ Plo_i,
    const _Float16* __restrict__ Phi_j, const _Float16* __restrict__ Plo_j,
    float* __restrict__ yi, float* __restrict__ yj, int ntiles_full)
{
  const int wave = threadIdx.x >> 5;
  const int lane = threadIdx.x & 31;
  const int tile = blockIdx.x * 8 + wave;
  if (tile >= ntiles_full) return;               // wave-uniform: EXEC stays all-ones
  const int row0 = tile << 4;
  const int r = lane & 15, hb = lane >> 4;

  const float* xrow = px + (long)(row0 + r) * FD;
  v16h a0h, a0l, a1h, a1l;
  make_frag_split_f32(xrow, 0, hb, a0h, a0l);
  make_frag_split_f32(xrow, 32, hb, a1h, a1l);

#pragma unroll
  for (int m = 0; m < 2; ++m) {
    const _Float16* Ph = m ? Phi_j : Phi_i;
    const _Float16* Pl = m ? Plo_j : Plo_i;
    float* y = m ? yj : yi;
#pragma unroll
    for (int nt = 0; nt < 4; ++nt) {
      const _Float16* ph = Ph + (nt * 16 + r) * FD;   // B lane holds out-col n = nt*16+r
      const _Float16* pl = Pl + (nt * 16 + r) * FD;
      v16h b0h, b1h, b0l, b1l;
      load_frag_f16(ph, 0, hb, b0h);
      load_frag_f16(ph, 32, hb, b1h);
      load_frag_f16(pl, 0, hb, b0l);
      load_frag_f16(pl, 32, hb, b1l);
      // Two independent accumulation chains (hi-path / correction-path).
      v8f acc0 = {};
      v8f acc1 = {};
      acc0 = WMMA_F16(a0h, b0h, acc0);
      acc1 = WMMA_F16(a0l, b0h, acc1);
      acc0 = WMMA_F16(a1h, b1h, acc0);
      acc1 = WMMA_F16(a1l, b1h, acc1);
      acc0 = WMMA_F16(a0h, b0l, acc0);
      acc1 = WMMA_F16(a1h, b1l, acc1);
      // D element (M = v + 8*hb, N = r) -> y[row0 + M][nt*16 + r]
      float* yp = y + (long)(row0 + 8 * hb) * FD + nt * 16 + r;
#pragma unroll
      for (int v = 0; v < 8; ++v) yp[(long)v * FD] = acc0[v] + acc1[v];
    }
  }
}

// ---------------------------------------------------------------------------
// Kernel 2b: scalar tail for nrows % 16 leftover rows (zero blocks for the
// reference shape; exists for generality). Reconstructs P ~ hi + lo.
// ---------------------------------------------------------------------------
__global__ __launch_bounds__(256) void node_gemm_tail_kernel(
    const float* __restrict__ px,
    const _Float16* __restrict__ Phi_i, const _Float16* __restrict__ Plo_i,
    const _Float16* __restrict__ Phi_j, const _Float16* __restrict__ Plo_j,
    float* __restrict__ yi, float* __restrict__ yj,
    int row_start, int nrows)
{
  const int nrem = nrows - row_start;
  for (int idx = threadIdx.x; idx < nrem * FD; idx += 256) {
    const int rr = idx >> 6, g = idx & 63;
    const float* x = px + (long)(row_start + rr) * FD;
    float ai = 0.f, aj = 0.f;
    for (int f = 0; f < FD; ++f) {
      const float pi = (float)Phi_i[g * FD + f] + (float)Plo_i[g * FD + f];
      const float pj = (float)Phi_j[g * FD + f] + (float)Plo_j[g * FD + f];
      ai = fmaf(x[f], pi, ai);
      aj = fmaf(x[f], pj, aj);
    }
    yi[(long)(row_start + rr) * FD + g] = ai;
    yj[(long)(row_start + rr) * FD + g] = aj;
  }
}

// ---------------------------------------------------------------------------
// Kernel 3: zero px_out region of d_out (harness poisons it with 0xAA).
// ---------------------------------------------------------------------------
__global__ void zero_kernel(v4f* __restrict__ p, long n4)
{
  const long i = (long)blockIdx.x * blockDim.x + threadIdx.x;
  if (i < n4) {
    v4f z = {0.f, 0.f, 0.f, 0.f};
    p[i] = z;
  }
}

// ---------------------------------------------------------------------------
// Kernel 4: edge pass. 16 lanes per (edge,axis) row, 4 floats per lane.
// ix = diff * (gather(y_i) + gather(y_j)); NT-store ix (614 MB stream, keep L2
// for the 77 MB y arrays); f32 atomic scatter-add into px_out (L2-resident).
// ---------------------------------------------------------------------------
__global__ __launch_bounds__(256) void edge_kernel(
    const int* __restrict__ idx_i, const int* __restrict__ idx_j,
    const float* __restrict__ diff,                 // [E*3]
    const float* __restrict__ yi, const float* __restrict__ yj,
    float* __restrict__ ix_out,                     // [E*3][64]
    float* __restrict__ px_out,                     // [N*3][64]
    int nrows)                                      // E*3
{
  const long t = (long)blockIdx.x * blockDim.x + threadIdx.x;
  const int row = (int)(t >> 4);
  if (row >= nrows) return;
  const int c = ((int)t & 15) * 4;
  const int e = row / 3;
  const int v = row - e * 3;

  const int ii = __ldg(idx_i + e);
  const int jj = __ldg(idx_j + e);
  const float s = __ldg(diff + row);

  const long ri = (long)(ii * 3 + v) * FD + c;
  const long rj = (long)(jj * 3 + v) * FD + c;
  const v4f a = *(const v4f*)(yi + ri);
  const v4f b = *(const v4f*)(yj + rj);
  const v4f o = s * (a + b);

  __builtin_nontemporal_store(o, (v4f*)(ix_out + (long)row * FD + c));

  float* pb = px_out + ri;
  atomicAdd(pb + 0, o.x);
  atomicAdd(pb + 1, o.y);
  atomicAdd(pb + 2, o.z);
  atomicAdd(pb + 3, o.w);
}

// ---------------------------------------------------------------------------
extern "C" void kernel_launch(void* const* d_in, const int* in_sizes, int n_in,
                              void* d_out, int out_size, void* d_ws, size_t ws_size,
                              hipStream_t stream) {
  const int*   idx_i = (const int*)d_in[0];
  const int*   idx_j = (const int*)d_in[1];
  const float* px    = (const float*)d_in[2];
  const float* diff  = (const float*)d_in[3];
  const float* W_pp  = (const float*)d_in[4];
  const float* W_i   = (const float*)d_in[5];
  const float* W_j   = (const float*)d_in[6];
  const float* W_ii  = (const float*)d_in[7];

  const int E          = in_sizes[0];
  const int node_rows  = in_sizes[2] / FD;   // N*3 = 150000
  const int edge_rows  = E * 3;              // 2.4M

  float* px_out = (float*)d_out;                                   // [node_rows][64]
  float* ix_out = (float*)d_out + (size_t)node_rows * FD;          // [edge_rows][64]

  // Workspace layout: 4x f16[64][64] weight splits, then y_i / y_j (fp32).
  char* ws = (char*)d_ws;
  _Float16* Phi_i = (_Float16*)ws;  ws += FD * FD * sizeof(_Float16);
  _Float16* Plo_i = (_Float16*)ws;  ws += FD * FD * sizeof(_Float16);
  _Float16* Phi_j = (_Float16*)ws;  ws += FD * FD * sizeof(_Float16);
  _Float16* Plo_j = (_Float16*)ws;  ws += FD * FD * sizeof(_Float16);
  float* yi = (float*)ws;           ws += (size_t)node_rows * FD * sizeof(float);
  float* yj = (float*)ws;

  // 1) Fold weights into P_i, P_j (f16 hi/lo splits).
  fold_weights_kernel<<<1, 256, 0, stream>>>(W_pp, W_i, W_j, W_ii,
                                             Phi_i, Plo_i, Phi_j, Plo_j);

  // 2) Node GEMM via WMMA over full 16-row tiles; scalar tail for remainder.
  const int ntiles_full = node_rows / 16;
  if (ntiles_full > 0) {
    const int gb = (ntiles_full + 7) / 8;
    node_gemm_kernel<<<gb, 256, 0, stream>>>(px, Phi_i, Plo_i, Phi_j, Plo_j,
                                             yi, yj, ntiles_full);
  }
  if (node_rows % 16) {
    node_gemm_tail_kernel<<<1, 256, 0, stream>>>(px, Phi_i, Plo_i, Phi_j, Plo_j,
                                                 yi, yj, ntiles_full * 16, node_rows);
  }

  // 3) Zero the px_out accumulation region.
  const long n4 = (long)node_rows * FD / 4;
  zero_kernel<<<(int)((n4 + 255) / 256), 256, 0, stream>>>((v4f*)px_out, n4);

  // 4) Edge pass: gather, scale, stream ix, scatter-add px_out.
  const long ethreads = (long)edge_rows * 16;
  edge_kernel<<<(int)((ethreads + 255) / 256), 256, 0, stream>>>(
      idx_i, idx_j, diff, yi, yj, ix_out, px_out, edge_rows);
}
